// VllmMixtureOfExpertsOp_67757404062156
// MI455X (gfx1250) — matrix operand order repaired and checked by reading, not compiled, not run on hardware
//
#include <hip/hip_runtime.h>
#include <math.h>

#define BT    256
#define HID   2048
#define NEXP  8
#define INTER 4096
#define TOPK  2
#define KCHUNK 128              // K elements staged per pipeline stage
#define RS    (KCHUNK + 4)      // padded LDS row stride (floats): conflict-free

typedef __attribute__((ext_vector_type(16))) __bf16 v16bf;
typedef __attribute__((ext_vector_type(8)))  float  v8f;
typedef __attribute__((ext_vector_type(4)))  float  f32x4;
typedef __attribute__((ext_vector_type(4)))  int    v4i;

#if defined(__has_builtin)
#if __has_builtin(__builtin_amdgcn_global_load_async_to_lds_b128)
#define HAVE_ASYNC 1
#endif
#endif

#ifdef HAVE_ASYNC
typedef __attribute__((address_space(1))) v4i as1_v4i;
typedef __attribute__((address_space(3))) v4i as3_v4i;
#endif

// Copy one 16-byte segment global -> LDS (async DMA when available).
__device__ __forceinline__ void stage_seg(const float* __restrict__ g,
                                          float* __restrict__ l) {
#ifdef HAVE_ASYNC
  __builtin_amdgcn_global_load_async_to_lds_b128((as1_v4i*)g, (as3_v4i*)l, 0, 0);
#else
  *(f32x4*)l = *(const f32x4*)g;
#endif
}

__device__ __forceinline__ void stage_fence() {
#ifdef HAVE_ASYNC
  asm volatile("s_wait_asynccnt 0x0" ::: "memory");
#endif
}

// ---------------------------------------------------------------------------
// WMMA fragment helpers (wave32, 16x16x32 bf16), f32 source converted on the
// fly (v_cvt_pk_bf16_f32; negligible vs. the 23.3 TB/s weight stream).
// A (16x32 MxK): lane L holds m = L%16; lanes 0-15 carry K runs [k0,k0+8) and
// [k0+16,k0+24); lanes 16-31 carry [k0+8,k0+16) and [k0+24,k0+32).
// ---------------------------------------------------------------------------
__device__ __forceinline__ v16bf load_a_frag(const float* __restrict__ row,
                                             int k0, int lane) {
  const int off = (lane & 16) ? 8 : 0;
  const f32x4* p0 = reinterpret_cast<const f32x4*>(row + k0 + off);
  const f32x4* p1 = reinterpret_cast<const f32x4*>(row + k0 + 16 + off);
  f32x4 x0 = p0[0], x1 = p0[1], y0 = p1[0], y1 = p1[1];
  v16bf r;
  r[0]=(__bf16)x0.x;  r[1]=(__bf16)x0.y;  r[2]=(__bf16)x0.z;  r[3]=(__bf16)x0.w;
  r[4]=(__bf16)x1.x;  r[5]=(__bf16)x1.y;  r[6]=(__bf16)x1.z;  r[7]=(__bf16)x1.w;
  r[8]=(__bf16)y0.x;  r[9]=(__bf16)y0.y;  r[10]=(__bf16)y0.z; r[11]=(__bf16)y0.w;
  r[12]=(__bf16)y1.x; r[13]=(__bf16)y1.y; r[14]=(__bf16)y1.z; r[15]=(__bf16)y1.w;
  return r;
}

// B (32x16 KxN) read from an LDS-staged tile: lane L holds n = L%16 (= row of
// the weight matrix); lanes 0-15 carry K [0,8) and [16,24), lanes 16-31 carry
// K [8,16) and [24,32) of WMMA step j within the staged KCHUNK.
__device__ __forceinline__ v16bf load_b_lds(const float* __restrict__ matbase,
                                            int j, int lane) {
  const int row = lane & 15;
  const int off = (lane & 16) ? 8 : 0;
  const f32x4* p0 = reinterpret_cast<const f32x4*>(matbase + row * RS + j * 32 + off);
  const f32x4* p1 = reinterpret_cast<const f32x4*>(matbase + row * RS + j * 32 + 16 + off);
  f32x4 x0 = p0[0], x1 = p0[1], x2 = p1[0], x3 = p1[1];
  v16bf r;
  r[0]=(__bf16)x0.x;  r[1]=(__bf16)x0.y;  r[2]=(__bf16)x0.z;  r[3]=(__bf16)x0.w;
  r[4]=(__bf16)x1.x;  r[5]=(__bf16)x1.y;  r[6]=(__bf16)x1.z;  r[7]=(__bf16)x1.w;
  r[8]=(__bf16)x2.x;  r[9]=(__bf16)x2.y;  r[10]=(__bf16)x2.z; r[11]=(__bf16)x2.w;
  r[12]=(__bf16)x3.x; r[13]=(__bf16)x3.y; r[14]=(__bf16)x3.z; r[15]=(__bf16)x3.w;
  return r;
}

#define WMMA_BF16(A, B, C) \
  __builtin_amdgcn_wmma_f32_16x16x32_bf16(false, (A), false, (B), (short)0, (C), false, false)

// ---------------------------------------------------------------------------
__global__ void moe_zero_out(float* __restrict__ out, int n) {
  int i = blockIdx.x * 256 + threadIdx.x;
  if (i < n) out[i] = 0.0f;
}

// ---------------------------------------------------------------------------
// Routing: compact per-expert token groups.  Last-wins dedupe matches the
// reference scatter (`.at[rows, sel].set(...)`) when a token lists the same
// expert twice.
// ---------------------------------------------------------------------------
__global__ void moe_route(const int*   __restrict__ routing,   // [BT, TOPK]
                          const float* __restrict__ rweights,  // [BT, TOPK]
                          int* counts, int* offsets, int* cursor,
                          int* toklist, float* wlist) {
  int t = threadIdx.x;
  if (t < NEXP) { counts[t] = 0; cursor[t] = 0; }
  __syncthreads();
  for (int k = 0; k < TOPK; ++k) {
    int e = routing[t * TOPK + k];
    bool keep = true;
    for (int k2 = k + 1; k2 < TOPK; ++k2)
      if (routing[t * TOPK + k2] == e) keep = false;   // last wins
    if (keep) atomicAdd(&counts[e], 1);
  }
  __syncthreads();
  if (t == 0) {
    int o = 0;
    for (int e = 0; e < NEXP; ++e) { offsets[e] = o; o += counts[e]; }
  }
  __syncthreads();
  for (int k = 0; k < TOPK; ++k) {
    int e = routing[t * TOPK + k];
    bool keep = true;
    for (int k2 = k + 1; k2 < TOPK; ++k2)
      if (routing[t * TOPK + k2] == e) keep = false;
    if (keep) {
      int pos = offsets[e] + atomicAdd(&cursor[e], 1);
      toklist[pos] = t;
      wlist[pos]   = rweights[t * TOPK + k];
    }
  }
}

// ---------------------------------------------------------------------------
// Kernel 2: act[slot, i] = silu(x @ w13[e, i, :]) * (x @ w13[e, INTER+i, :]).
// Grid = E * (INTER/16).  The 16(up)+16(gate) weight rows for a KCHUNK are
// async-DMA'd into a double-buffered LDS tile once per block; all 8 waves
// consume it.  Each wave owns two 16-token M-tiles (covers max 16 tiles).
// ---------------------------------------------------------------------------
__global__ void __launch_bounds__(256)
moe_up_gate(const float* __restrict__ hidden,
            const float* __restrict__ w13,
            const int*   __restrict__ counts,
            const int*   __restrict__ offsets,
            const int*   __restrict__ toklist,
            float*       __restrict__ act) {
  const int bx    = blockIdx.x;
  const int e     = bx >> 8;          // INTER/16 = 256 n-tiles per expert
  const int n0    = (bx & 255) * 16;
  const int cnt   = counts[e];
  if (cnt == 0) return;
  const int base   = offsets[e];
  const int mtiles = (cnt + 15) >> 4;
  const int wave   = threadIdx.x >> 5;
  const int lane   = threadIdx.x & 31;

  __shared__ float lds[2][2][16 * RS];          // [buf][up|gate][row*RS + k]

  const float* wbase = w13 + (size_t)e * (2 * INTER) * HID;

  // A-row pointers for this wave's two M-tiles
  const int m_in_lane = lane & 15;
  int mt0 = wave, mt1 = wave + 8;
  const float* arow0 = hidden;
  const float* arow1 = hidden;
  if (mt0 < mtiles) {
    int idx = mt0 * 16 + m_in_lane; if (idx >= cnt) idx = 0;
    arow0 = hidden + (size_t)toklist[base + idx] * HID;
  }
  if (mt1 < mtiles) {
    int idx = mt1 * 16 + m_in_lane; if (idx >= cnt) idx = 0;
    arow1 = hidden + (size_t)toklist[base + idx] * HID;
  }
  v8f au0 = {}, ag0 = {}, au1 = {}, ag1 = {};

  // stage one KCHUNK (up + gate tiles): 1024 16B segments, 4 per lane-instr
  auto stage = [&](int buf, int kbase) {
    for (int i = 0; i < 4; ++i) {
      int seg = (((wave << 2) + i) << 5) | lane;   // 0..1023
      int mat = seg >> 9;
      int rem = seg & 511;
      int row = rem >> 5;
      int ksg = rem & 31;                          // 16B unit within row
      const float* g = wbase + (size_t)(mat * INTER + n0 + row) * HID
                             + kbase + ksg * 4;
      stage_seg(g, &lds[buf][mat][row * RS + ksg * 4]);
    }
  };

  stage(0, 0);
  stage_fence();
  __syncthreads();

  const int NST = HID / KCHUNK;                    // 16 stages
  for (int st = 0; st < NST; ++st) {
    const int cur = st & 1;
    if (st + 1 < NST) stage(cur ^ 1, (st + 1) * KCHUNK);

    for (int j = 0; j < 4; ++j) {                  // 4 WMMA K-steps per stage
      int kb = st * KCHUNK + j * 32;
      v16bf bu = load_b_lds(&lds[cur][0][0], j, lane);
      v16bf bg = load_b_lds(&lds[cur][1][0], j, lane);
      if (mt0 < mtiles) {
        v16bf a = load_a_frag(arow0, kb, lane);
        au0 = WMMA_BF16(a, bu, au0);
        ag0 = WMMA_BF16(a, bg, ag0);
      }
      if (mt1 < mtiles) {
        v16bf a = load_a_frag(arow1, kb, lane);
        au1 = WMMA_BF16(a, bu, au1);
        ag1 = WMMA_BF16(a, bg, ag1);
      }
    }
    if (st + 1 < NST) stage_fence();
    __syncthreads();
  }

  // epilogue: silu(up)*gate  (v_rcp instead of the IEEE divide macro)
  const int n     = lane & 15;
  const int mhalf = (lane & 16) ? 8 : 0;
  auto epi = [&](int mt, v8f& au, v8f& ag) {
    if (mt >= mtiles) return;
    for (int r = 0; r < 8; ++r) {
      float u = au[r], g = ag[r];
      float s = u * __builtin_amdgcn_rcpf(1.0f + __expf(-u));
      int m = mt * 16 + r + mhalf;
      if (m < cnt) act[(size_t)(base + m) * INTER + n0 + n] = s * g;
    }
  };
  epi(mt0, au0, ag0);
  epi(mt1, au1, ag1);
}

// ---------------------------------------------------------------------------
// Kernel 3: out[t, h] += rw * (act[slot, :] @ w2[e, h, :]).  Grid = E*(HID/16).
// Same LDS-staged weight pipeline (one matrix).  Exactly <=2 commutative f32
// atomic adds per output element -> deterministic.
// ---------------------------------------------------------------------------
__global__ void __launch_bounds__(256)
moe_down(const float* __restrict__ act,
         const float* __restrict__ w2,
         const int*   __restrict__ counts,
         const int*   __restrict__ offsets,
         const int*   __restrict__ toklist,
         const float* __restrict__ wlist,
         float*       __restrict__ out) {
  const int bx    = blockIdx.x;
  const int e     = bx >> 7;          // HID/16 = 128 n-tiles per expert
  const int n0    = (bx & 127) * 16;
  const int cnt   = counts[e];
  if (cnt == 0) return;
  const int base   = offsets[e];
  const int mtiles = (cnt + 15) >> 4;
  const int wave   = threadIdx.x >> 5;
  const int lane   = threadIdx.x & 31;

  __shared__ float lds[2][16 * RS];

  const float* wbase = w2 + (size_t)e * HID * INTER;

  const int m_in_lane = lane & 15;
  int mt0 = wave, mt1 = wave + 8;
  const float* arow0 = act;
  const float* arow1 = act;
  if (mt0 < mtiles) {
    int idx = mt0 * 16 + m_in_lane; if (idx >= cnt) idx = 0;
    arow0 = act + (size_t)(base + idx) * INTER;
  }
  if (mt1 < mtiles) {
    int idx = mt1 * 16 + m_in_lane; if (idx >= cnt) idx = 0;
    arow1 = act + (size_t)(base + idx) * INTER;
  }
  v8f acc0 = {}, acc1 = {};

  auto stage = [&](int buf, int kbase) {
    for (int i = 0; i < 2; ++i) {                  // 512 segs / 8 waves
      int seg = (((wave << 1) + i) << 5) | lane;   // 0..511
      int row = seg >> 5;
      int ksg = seg & 31;
      const float* g = wbase + (size_t)(n0 + row) * INTER + kbase + ksg * 4;
      stage_seg(g, &lds[buf][row * RS + ksg * 4]);
    }
  };

  stage(0, 0);
  stage_fence();
  __syncthreads();

  const int NST = INTER / KCHUNK;                  // 32 stages
  for (int st = 0; st < NST; ++st) {
    const int cur = st & 1;
    if (st + 1 < NST) stage(cur ^ 1, (st + 1) * KCHUNK);

    for (int j = 0; j < 4; ++j) {
      int kb = st * KCHUNK + j * 32;
      v16bf b = load_b_lds(&lds[cur][0], j, lane);
      if (mt0 < mtiles) {
        v16bf a = load_a_frag(arow0, kb, lane);
        acc0 = WMMA_BF16(a, b, acc0);
      }
      if (mt1 < mtiles) {
        v16bf a = load_a_frag(arow1, kb, lane);
        acc1 = WMMA_BF16(a, b, acc1);
      }
    }
    if (st + 1 < NST) stage_fence();
    __syncthreads();
  }

  const int n     = lane & 15;
  const int mhalf = (lane & 16) ? 8 : 0;
  auto epi = [&](int mt, v8f& acc) {
    if (mt >= mtiles) return;
    for (int r = 0; r < 8; ++r) {
      int m = mt * 16 + r + mhalf;
      if (m < cnt) {
        int   t = toklist[base + m];
        float w = wlist[base + m];
        unsafeAtomicAdd(&out[(size_t)t * HID + n0 + n], acc[r] * w);
      }
    }
  };
  epi(mt0, acc0);
  epi(mt1, acc1);
}

// ---------------------------------------------------------------------------
extern "C" void kernel_launch(void* const* d_in, const int* in_sizes, int n_in,
                              void* d_out, int out_size, void* d_ws, size_t ws_size,
                              hipStream_t stream) {
  const float* hidden   = (const float*)d_in[0];
  const int*   routing  = (const int*)  d_in[1];
  const float* rweights = (const float*)d_in[2];
  const float* w13      = (const float*)d_in[3];
  const float* w2       = (const float*)d_in[4];
  float*       out      = (float*)d_out;

  // workspace (elements): counts[8] | offsets[8] | cursor[8] | pad |
  // toklist[512] | wlist[512] | pad | act[512*INTER]  (~8 MB total)
  int*   counts  = (int*)d_ws;
  int*   offsets = counts + 8;
  int*   cursor  = offsets + 8;
  int*   toklist = (int*)d_ws + 32;
  float* wlist   = (float*)d_ws + 32 + 512;
  float* act     = (float*)d_ws + 1280;

  moe_zero_out<<<(out_size + 255) / 256, 256, 0, stream>>>(out, out_size);
  moe_route<<<1, BT, 0, stream>>>(routing, rweights, counts, offsets, cursor,
                                  toklist, wlist);
  moe_up_gate<<<NEXP * (INTER / 16), 256, 0, stream>>>(hidden, w13, counts,
                                                       offsets, toklist, act);
  moe_down<<<NEXP * (HID / 16), 256, 0, stream>>>(act, w2, counts, offsets,
                                                  toklist, wlist, out);
}
